// AttentionAggregator_f_2551210574178
// MI455X (gfx1250) — matrix operand order repaired and checked by reading, not compiled, not run on hardware
//
#include <hip/hip_runtime.h>
#include <hip/hip_bf16.h>

#define DIMK 512   // IN_DIM == OUT_DIM == 512
#define SLOPE 0.1f

typedef __bf16 bf16;
typedef bf16  v16bf __attribute__((ext_vector_type(16)));
typedef bf16  v8bf  __attribute__((ext_vector_type(8)));
typedef float v8f   __attribute__((ext_vector_type(8)));
typedef int   v4i   __attribute__((ext_vector_type(4)));
typedef int   v8i   __attribute__((ext_vector_type(8)));

__device__ __forceinline__ void atomicAddF(float* p, float v) {
    __hip_atomic_fetch_add(p, v, __ATOMIC_RELAXED, __HIP_MEMORY_SCOPE_AGENT);
}

// ---------------------------------------------------------------------------
// Convert features (f32) -> bf16, elementwise.
__global__ __launch_bounds__(256) void cvt_feat_kernel(const float* __restrict__ f,
                                                       bf16* __restrict__ o, size_t cnt) {
    size_t i = (size_t)blockIdx.x * blockDim.x + threadIdx.x;
    if (i < cnt) o[i] = (bf16)f[i];
}

// Transpose W (512x512 f32, row-major [k][n]) -> Wt bf16 [n][k].
__global__ __launch_bounds__(256) void cvt_wt_kernel(const float* __restrict__ W,
                                                     bf16* __restrict__ Wt) {
    int idx = blockIdx.x * blockDim.x + threadIdx.x;   // 512*512 threads
    int k = idx >> 9;
    int n = idx & 511;
    Wt[(size_t)n * DIMK + k] = (bf16)W[(size_t)k * DIMK + n];
}

// ---------------------------------------------------------------------------
// emb = featb(bf16) @ Wt(bf16)^T + bias, fp32 accumulate via WMMA.
// The block's 16-row A slab (16x512 bf16 = 16 KB) is DMA'd into LDS once via
// the Tensor Data Mover (tensor_load_to_lds, TENSORcnt); all 4 waves then feed
// their WMMAs from LDS while streaming B tiles from global (L2-resident).
__global__ __launch_bounds__(128) void gemm_wmma_kernel(const bf16* __restrict__ A,
                                                        const bf16* __restrict__ Bt,
                                                        const float* __restrict__ bias,
                                                        float* __restrict__ C, int n) {
    __shared__ bf16 lds_a[16 * DIMK];          // 16 KB A tile, row-major [m][k]

    const int lane  = threadIdx.x & 31;
    const int wave  = threadIdx.x >> 5;
    const int tileM = blockIdx.x * 16;
    const int tileN = (blockIdx.y * 4 + wave) * 16;
    const int half  = lane >> 4;    // 0: lanes 0-15, 1: lanes 16-31
    const int l16   = lane & 15;

    if (wave == 0) {
        // Tensor DMA descriptor (CDNA5 ISA):
        unsigned lds_off = (unsigned)(unsigned long long)(&lds_a[0]);   // addr[31:0] = LDS offset
        unsigned long long ga = (unsigned long long)(A + (size_t)tileM * DIMK);
        // Group 0: count=1 | lds_addr | global_addr[56:0] | type=2
        v4i g0 = { 1,
                   (int)lds_off,
                   (int)(unsigned)(ga & 0xFFFFFFFFu),
                   (int)((unsigned)((ga >> 32) & 0x1FFFFFFu) | (2u << 30)) };
        // Group 1: mask=0, data_size=1 (2B), dim0=512, dim1=n, tile0=512, tile1=16,
        //          dim0_stride=512, no padding/iterate/barrier.
        v8i g1 = { (int)(1u << 16),                                   // data_size=2B
                   (int)(((unsigned)DIMK & 0xFFFFu) << 16),           // tensor_dim0 lo
                   (int)((((unsigned)DIMK >> 16) & 0xFFFFu) |
                         (((unsigned)n & 0xFFFFu) << 16)),            // dim0 hi | dim1 lo
                   (int)((((unsigned)n >> 16) & 0xFFFFu) |
                         ((unsigned)DIMK << 16)),                     // dim1 hi | tile_dim0
                   16,                                                // tile_dim1=16, tile_dim2=0
                   DIMK,                                              // dim0_stride lo (elements)
                   0,                                                 // dim0_stride hi | dim1_stride lo
                   0 };
        asm volatile("tensor_load_to_lds %0, %1" :: "s"(g0), "s"(g1) : "memory");
        __builtin_amdgcn_s_wait_tensorcnt(0);
    }
    __syncthreads();

    const int bcol = tileN + l16;
    // A from LDS (contiguous K, matching B): half 0 -> K 0..15; half 1 -> K 16..31
    const bf16* ap = &lds_a[(size_t)l16 * DIMK + half * 16];
    // B (column-major load of Wt row): half 0 -> K 0..15; half 1 -> K 16..31
    const bf16* bp = Bt + (size_t)bcol * DIMK + half * 16;

    v8f acc = {};
#pragma unroll
    for (int k = 0; k < DIMK; k += 32) {
        v8bf a0 = *(const v8bf*)(ap + k);          // ds_load_b128
        v8bf a1 = *(const v8bf*)(ap + k + 8);
        v16bf av = __builtin_shufflevector(a0, a1, 0,1,2,3,4,5,6,7,8,9,10,11,12,13,14,15);
        v16bf bv = *(const v16bf*)(bp + k);        // 2x global_load_b128
        acc = __builtin_amdgcn_wmma_f32_16x16x32_bf16(false, av, false, bv,
                                                      (short)0, acc, false, false);
    }

    const float bb = bias[bcol];
#pragma unroll
    for (int i = 0; i < 8; ++i) {
        int row = tileM + i + half * 8;     // C/D layout: VGPR i = row i (+8 for upper half)
        if (row < n) C[(size_t)row * DIMK + bcol] = acc[i] + bb;
    }
}

// ---------------------------------------------------------------------------
// s[i] = dot(emb[i], a[0:512]); t[i] = dot(emb[i], a[512:1024]). One wave per row.
__global__ __launch_bounds__(256) void scores_kernel(const float* __restrict__ emb,
                                                     const float* __restrict__ a,
                                                     float* __restrict__ s,
                                                     float* __restrict__ t, int n) {
    int row  = blockIdx.x * (blockDim.x >> 5) + (threadIdx.x >> 5);
    int lane = threadIdx.x & 31;
    if (row >= n) return;
    const float* er = emb + (size_t)row * DIMK;
    float ss = 0.f, tt = 0.f;
#pragma unroll 4
    for (int j = lane; j < DIMK; j += 32) {
        float v = er[j];
        ss += v * a[j];
        tt += v * a[DIMK + j];
    }
#pragma unroll
    for (int off = 16; off > 0; off >>= 1) {
        ss += __shfl_down(ss, off, 32);
        tt += __shfl_down(tt, off, 32);
    }
    if (lane == 0) { s[row] = ss; t[row] = tt; }
}

// ---------------------------------------------------------------------------
// Zero the accumulation buffers (d_out is poisoned / left divided from prior call).
__global__ __launch_bounds__(256) void zero_kernel(float* __restrict__ out,
                                                   float* __restrict__ rsum,
                                                   size_t nout, int n) {
    size_t i = (size_t)blockIdx.x * blockDim.x + threadIdx.x;
    if (i < nout) out[i] = 0.f;
    if (i < (size_t)n) rsum[i] = 0.f;
}

// ---------------------------------------------------------------------------
// Edge pass 1: per-edge weight + row_sum scatter (scalar f32 atomics, L2-resident).
__global__ __launch_bounds__(256) void edge_weight_kernel(const int* __restrict__ edges,
                                                          const float* __restrict__ s,
                                                          const float* __restrict__ t,
                                                          float* __restrict__ ew,
                                                          float* __restrict__ rsum, int E) {
    int e = blockIdx.x * blockDim.x + threadIdx.x;
    if (e >= E) return;
    int src = edges[2 * e];
    int dst = edges[2 * e + 1];
    float x  = s[src] + t[dst];
    float lr = x > 0.f ? x : SLOPE * x;
    float w  = expf(lr);
    ew[e] = w;
    atomicAddF(&rsum[src], w);
}

// ---------------------------------------------------------------------------
// Edge pass 2: out[src,:] += w * emb[dst,:]. 128 threads/edge, float4 gathers,
// f32 atomic scatter (no-return -> global_atomic_add_f32, resolves in L2).
__global__ __launch_bounds__(256) void edge_scatter_kernel(const int* __restrict__ edges,
                                                           const float* __restrict__ ew,
                                                           const float* __restrict__ emb,
                                                           float* __restrict__ out, int E) {
    long long gid = (long long)blockIdx.x * blockDim.x + threadIdx.x;
    int e = (int)(gid >> 7);
    if (e >= E) return;
    int chunk = (int)(gid & 127);              // 128 chunks x float4 = 512 floats
    int src = edges[2 * e];
    int dst = edges[2 * e + 1];
    float w = ew[e];
    float4 v = ((const float4*)(emb + (size_t)dst * DIMK))[chunk];
    float* op = out + (size_t)src * DIMK + chunk * 4;
    atomicAddF(op + 0, w * v.x);
    atomicAddF(op + 1, w * v.y);
    atomicAddF(op + 2, w * v.z);
    atomicAddF(op + 3, w * v.w);
}

// ---------------------------------------------------------------------------
// Final normalization: out[i,:] /= row_sum[i].
__global__ __launch_bounds__(256) void divide_kernel(float* __restrict__ out,
                                                     const float* __restrict__ rsum,
                                                     size_t cnt) {
    size_t i = (size_t)blockIdx.x * blockDim.x + threadIdx.x;
    if (i < cnt) out[i] = out[i] / rsum[i >> 9];
}

// ---------------------------------------------------------------------------
extern "C" void kernel_launch(void* const* d_in, const int* in_sizes, int n_in,
                              void* d_out, int out_size, void* d_ws, size_t ws_size,
                              hipStream_t stream) {
    const float* features = (const float*)d_in[0];
    const int*   edges    = (const int*)  d_in[1];
    const float* W        = (const float*)d_in[2];
    const float* bias     = (const float*)d_in[3];
    const float* a        = (const float*)d_in[4];

    const int n = in_sizes[0] / DIMK;      // 40000 nodes
    const int E = in_sizes[1] / 2;         // 680640 edges
    float* out = (float*)d_out;

    // Workspace carve-up (256B aligned): ~126 MB total.
    char* ws = (char*)d_ws;
    size_t off = 0;
    auto take = [&](size_t bytes) -> void* {
        void* p = ws + off;
        off = (off + bytes + 255) & ~(size_t)255;
        return p;
    };
    bf16*  featb = (bf16*) take((size_t)n * DIMK * sizeof(bf16));
    bf16*  wtb   = (bf16*) take((size_t)DIMK * DIMK * sizeof(bf16));
    float* emb   = (float*)take((size_t)n * DIMK * sizeof(float));
    float* s     = (float*)take((size_t)n * sizeof(float));
    float* t     = (float*)take((size_t)n * sizeof(float));
    float* rsum  = (float*)take((size_t)n * sizeof(float));
    float* ew    = (float*)take((size_t)E * sizeof(float));

    const size_t nfeat = (size_t)n * DIMK;

    // 1. features -> bf16
    cvt_feat_kernel<<<(unsigned)((nfeat + 255) / 256), 256, 0, stream>>>(features, featb, nfeat);
    // 2. W -> Wt (bf16, transposed)
    cvt_wt_kernel<<<(DIMK * DIMK) / 256, 256, 0, stream>>>(W, wtb);
    // 3. WMMA GEMM: emb = featb @ W + b  (TDM-staged A tile in LDS)
    {
        dim3 grid((n + 15) / 16, DIMK / (16 * 4));   // (2500, 8)
        gemm_wmma_kernel<<<grid, 128, 0, stream>>>(featb, wtb, bias, emb, n);
    }
    // 4. node scores
    scores_kernel<<<(n + 7) / 8, 256, 0, stream>>>(emb, a, s, t, n);
    // 5. zero accumulators
    zero_kernel<<<(unsigned)((nfeat + 255) / 256), 256, 0, stream>>>(out, rsum, nfeat, n);
    // 6. edge weights + row sums
    edge_weight_kernel<<<(E + 255) / 256, 256, 0, stream>>>(edges, s, t, ew, rsum, E);
    // 7. weighted scatter
    {
        long long total = (long long)E * 128;
        edge_scatter_kernel<<<(unsigned)((total + 255) / 256), 256, 0, stream>>>(edges, ew, emb, out, E);
    }
    // 8. normalize
    divide_kernel<<<(unsigned)((nfeat + 255) / 256), 256, 0, stream>>>(out, rsum, nfeat);
}
